// DistributedImprovedDilatedAttention_61727269978796
// MI455X (gfx1250) — matrix operand order repaired and checked by reading, not compiled, not run on hardware
//
#include <hip/hip_runtime.h>

// MI455X / gfx1250: wave32, WMMA bf16 16x16x32, f32 accumulate.
typedef __attribute__((ext_vector_type(16))) __bf16 v16bf;
typedef __attribute__((ext_vector_type(2)))  __bf16 v2bf;
typedef __attribute__((ext_vector_type(8)))  float  v8f;

#define STR 72   // padded LDS row stride (elements) to stagger banks

union AFrag { v16bf v; unsigned int u[8]; };
union Acc   { v8f   v; float        f[8]; };

// f32 pair -> packed bf16x2 (hardware v_cvt_pk_bf16_f32 when available)
__device__ __forceinline__ unsigned int pack2bf(float a, float b) {
#if __has_builtin(__builtin_amdgcn_cvt_pk_bf16_f32)
    union { v2bf v; unsigned int u; } c;
    c.v = __builtin_amdgcn_cvt_pk_bf16_f32(a, b);
    return c.u;
#else
    union { __bf16 h[2]; unsigned int u; } c;
    c.h[0] = (__bf16)a; c.h[1] = (__bf16)b;
    return c.u;
#endif
}

__device__ __forceinline__ unsigned short f2bf(float f) {
    return (unsigned short)(pack2bf(f, 0.f) & 0xFFFFu);
}

__global__ __launch_bounds__(128)
void dilated_flash_attn_kernel(const float* __restrict__ Q,
                               const float* __restrict__ K,
                               const float* __restrict__ V,
                               const int*   __restrict__ pcausal,
                               float*       __restrict__ O)
{
    __shared__ unsigned short Qs [64 * STR];      // [qrow][d]   bf16
    __shared__ unsigned short Ks [64 * STR];      // [key ][d]   bf16
    __shared__ unsigned short VTs[64 * STR];      // [d   ][key] bf16 (transposed)
    __shared__ unsigned short Ps [4 * 16 * STR];  // per-wave P staging [row][key]

    const int tid  = threadIdx.x;
    const int lane = tid & 31;
    const int wave = tid >> 5;
    const int half = (lane >> 4) & 1;
    const int lcol = lane & 15;

    // ---- decode (group, batch, segment, head, query-tile) from blockIdx ----
    int tile = blockIdx.x;
    int qt   = tile & 15;          // query tile (64 rows each, m=1024)
    int blk  = tile >> 4;          // 0..55
    int grp, bl;
    if      (blk < 32) { grp = 0; bl = blk;      }
    else if (blk < 48) { grp = 1; bl = blk - 32; }
    else               { grp = 2; bl = blk - 48; }
    const int scnt  = 4 >> grp;            // segments: 4,2,1
    const int gseg  = 1024 << grp;         // 1024,2048,4096
    const int rate  = 1 << grp;            // 1,2,4
    const int off   = grp;                 // offset = grp % rate == grp
    const int seg   = bl % scnt;
    const int hl    = (bl / scnt) & 3;
    const int batch = bl / (scnt * 4);
    const int head  = grp * 4 + hl;
    const int causal = pcausal[0];

    // element address of dilated row j, dim d:
    //   addr = rowBase + j*rowStride + d   (layout [b, n, 12, 64], n-stride 768)
    const long long rowBase   = ((long long)batch * 4096 + (long long)seg * gseg + off) * 768
                              + (long long)head * 64;
    const long long rowStride = (long long)rate * 768;

    // per-thread fixed (row, d0) slot for cooperative loads
    // (each thread does 8 float4 loads: idx = tid + it*128)

    // ---- load Q tile (64 rows x 64 d) -> LDS bf16 ----
    for (int it = 0; it < 8; ++it) {
        int idx = tid + it * 128;       // 1024 float4 slots
        int row = idx >> 4;
        int d0  = (idx & 15) * 4;
        int j   = qt * 64 + row;
        const float4 qv = *(const float4*)(Q + rowBase + (long long)j * rowStride + d0);
        *(unsigned int*)&Qs[row * STR + d0]     = pack2bf(qv.x, qv.y);
        *(unsigned int*)&Qs[row * STR + d0 + 2] = pack2bf(qv.z, qv.w);
    }
    __syncthreads();

    // ---- Q A-fragments (16x32 bf16, per ISA A layout), fixed for all chunks ----
    AFrag qa[2];
    {
        int mrowq = wave * 16 + lcol;
        for (int f = 0; f < 2; ++f) {
            int dbase = f * 32;
            for (int v = 0; v < 8; ++v) {
                int k0 = dbase + ((v >= 4) ? 16 : 0) + half * 8 + 2 * (v & 3);
                qa[f].u[v] = *(const unsigned int*)&Qs[mrowq * STR + k0];
            }
        }
    }

    // ---- online-softmax state (C layout: row = i + 8*half, col = lcol) ----
    float mrow[8], lrow[8];
    Acc o[4];
    for (int i = 0; i < 8; ++i) { mrow[i] = -1e30f; lrow[i] = 0.f; }
    for (int n = 0; n < 4; ++n)
        for (int i = 0; i < 8; ++i) o[n].f[i] = 0.f;

    unsigned short* PW = &Ps[wave * 16 * STR];
    const float scaleS = 0.125f;   // 1/sqrt(64)
    const int qrow0 = qt * 64 + wave * 16 + half * 8;

    for (int ch = 0; ch < 16; ++ch) {      // 16 chunks x 64 keys = 1024 keys
        __syncthreads();
        // ---- load K chunk row-major, V chunk transposed -> LDS bf16 ----
        for (int it = 0; it < 8; ++it) {
            int idx = tid + it * 128;
            int row = idx >> 4;
            int d0  = (idx & 15) * 4;
            int j   = ch * 64 + row;
            long long base = rowBase + (long long)j * rowStride + d0;
            const float4 kv = *(const float4*)(K + base);
            *(unsigned int*)&Ks[row * STR + d0]     = pack2bf(kv.x, kv.y);
            *(unsigned int*)&Ks[row * STR + d0 + 2] = pack2bf(kv.z, kv.w);
            const float4 vv = *(const float4*)(V + base);
            VTs[(d0 + 0) * STR + row] = f2bf(vv.x);
            VTs[(d0 + 1) * STR + row] = f2bf(vv.y);
            VTs[(d0 + 2) * STR + row] = f2bf(vv.z);
            VTs[(d0 + 3) * STR + row] = f2bf(vv.w);
        }
        // prefetch next chunk's K/V tile toward the WGP (global_prefetch_b8)
        if (ch < 15) {
            int row = tid >> 4;
            int d0  = (tid & 15) * 4;
            long long nbase = rowBase + (long long)((ch + 1) * 64 + row) * rowStride + d0;
            __builtin_prefetch(K + nbase, 0, 3);
            __builtin_prefetch(V + nbase, 0, 3);
        }
        __syncthreads();

        // ---- S = Q K^T : 16 queries x 64 keys per wave, 8 WMMAs ----
        Acc s[4];
        for (int n = 0; n < 4; ++n) {
            AFrag kb0, kb1;
            int ncol = n * 16 + lcol;       // key column (B-matrix N)
            for (int v = 0; v < 8; ++v) {
                int k0 = ((v >= 4) ? 16 : 0) + half * 8 + 2 * (v & 3); // B layout: same K order as A
                kb0.u[v] = *(const unsigned int*)&Ks[ncol * STR + k0];
                kb1.u[v] = *(const unsigned int*)&Ks[ncol * STR + 32 + k0];
            }
            v8f z = {0.f, 0.f, 0.f, 0.f, 0.f, 0.f, 0.f, 0.f};
            v8f acc = __builtin_amdgcn_wmma_f32_16x16x32_bf16(
                false, qa[0].v, false, kb0.v, (short)0, z, false, false);
            acc = __builtin_amdgcn_wmma_f32_16x16x32_bf16(
                false, qa[1].v, false, kb1.v, (short)0, acc, false, false);
            s[n].v = acc;
        }

        // ---- scale + causal mask + per-row max (reduce over 16-lane half) ----
        float pm[8];
        for (int i = 0; i < 8; ++i) pm[i] = -1e30f;
        for (int n = 0; n < 4; ++n) {
            int kIdx = ch * 64 + n * 16 + lcol;
            for (int i = 0; i < 8; ++i) {
                float sv = s[n].f[i] * scaleS;
                if (causal && (kIdx > (qrow0 + i))) sv = -1e30f;
                s[n].f[i] = sv;
                pm[i] = fmaxf(pm[i], sv);
            }
        }
        for (int d = 1; d < 16; d <<= 1)
            for (int i = 0; i < 8; ++i)
                pm[i] = fmaxf(pm[i], __shfl_xor(pm[i], d, 32));

        float scl[8], rs[8];
        for (int i = 0; i < 8; ++i) {
            float mn = fmaxf(mrow[i], pm[i]);
            scl[i] = __expf(mrow[i] - mn);
            mrow[i] = mn;
            rs[i] = 0.f;
        }

        // ---- P = exp(S - m); stage to per-wave LDS (C layout -> row major) ----
        for (int n = 0; n < 4; ++n)
            for (int i = 0; i < 8; ++i) {
                float p = __expf(s[n].f[i] - mrow[i]);
                rs[i] += p;
                PW[(i + half * 8) * STR + n * 16 + lcol] = f2bf(p);
            }
        for (int d = 1; d < 16; d <<= 1)
            for (int i = 0; i < 8; ++i)
                rs[i] += __shfl_xor(rs[i], d, 32);
        for (int i = 0; i < 8; ++i) lrow[i] = lrow[i] * scl[i] + rs[i];
        for (int n = 0; n < 4; ++n)
            for (int i = 0; i < 8; ++i) o[n].f[i] *= scl[i];

        // ---- P A-fragments (LDS in-order per wave: store->load safe) ----
        AFrag pa0, pa1;
        for (int v = 0; v < 8; ++v) {
            int k0 = ((v >= 4) ? 16 : 0) + half * 8 + 2 * (v & 3);
            pa0.u[v] = *(const unsigned int*)&PW[lcol * STR + k0];
            pa1.u[v] = *(const unsigned int*)&PW[lcol * STR + 32 + k0];
        }
        // ---- O += P V : 8 WMMAs ----
        for (int n = 0; n < 4; ++n) {
            AFrag vb0, vb1;
            int ncol = n * 16 + lcol;       // output d column
            for (int v = 0; v < 8; ++v) {
                int k0 = ((v >= 4) ? 16 : 0) + half * 8 + 2 * (v & 3); // key pairs (same K order as A)
                vb0.u[v] = *(const unsigned int*)&VTs[ncol * STR + k0];
                vb1.u[v] = *(const unsigned int*)&VTs[ncol * STR + 32 + k0];
            }
            v8f acc = __builtin_amdgcn_wmma_f32_16x16x32_bf16(
                false, pa0.v, false, vb0.v, (short)0, o[n].v, false, false);
            acc = __builtin_amdgcn_wmma_f32_16x16x32_bf16(
                false, pa1.v, false, vb1.v, (short)0, acc, false, false);
            o[n].v = acc;
        }
    }

    // ---- finalize: O / l, scatter back to dilated positions (f32) ----
    for (int i = 0; i < 8; ++i) {
        float inv = (lrow[i] > 0.f) ? (1.0f / lrow[i]) : 0.f;
        int j = qt * 64 + wave * 16 + half * 8 + i;
        long long base = rowBase + (long long)j * rowStride;
        for (int n = 0; n < 4; ++n)
            O[base + n * 16 + lcol] = o[n].f[i] * inv;
    }
}

__global__ void zero_out_kernel(float* __restrict__ p, int nElem) {
    int i = blockIdx.x * blockDim.x + threadIdx.x;
    if (i < nElem) p[i] = 0.f;
}

extern "C" void kernel_launch(void* const* d_in, const int* in_sizes, int n_in,
                              void* d_out, int out_size, void* d_ws, size_t ws_size,
                              hipStream_t stream) {
    const float* Q  = (const float*)d_in[0];
    const float* K  = (const float*)d_in[1];
    const float* V  = (const float*)d_in[2];
    const int*   pc = (const int*)d_in[3];
    float* O = (float*)d_out;

    // LongNet semantics: un-attended (non-dilated) positions are zero.
    zero_out_kernel<<<(out_size + 255) / 256, 256, 0, stream>>>(O, out_size);

    // 56 attention blocks x 16 query tiles (64 rows) = 896 workgroups,
    // 128 threads (4 wave32) each.
    dilated_flash_attn_kernel<<<896, 128, 0, stream>>>(Q, K, V, pc, O);
}
